// TextAttention_20976620273958
// MI455X (gfx1250) — compile-verified, matrix operand order checked by reading, not run on hardware
//
#include <hip/hip_runtime.h>

typedef __attribute__((ext_vector_type(16))) _Float16 v16h;
typedef __attribute__((ext_vector_type(8)))  _Float16 v8h;
typedef __attribute__((ext_vector_type(8)))  float    v8f;
typedef __attribute__((ext_vector_type(4)))  int      v4i;

#define AS1_ __attribute__((address_space(1)))
#define AS3_ __attribute__((address_space(3)))

#define B_   2
#define L_   2048
#define HS_  2560
#define H_   16
#define HKV_ 8
#define DH_  256
#define SOFTCAP_ 50.0f
#define EPS_ 1e-6f
#define SCALING_ 1.0f

// ---- CDNA5 async global->LDS copy (ASYNCcnt) with compile-safe fallback ----
#if defined(__has_builtin)
#  if __has_builtin(__builtin_amdgcn_global_load_async_to_lds_b128) && \
      __has_builtin(__builtin_amdgcn_s_wait_asynccnt)
#    define USE_ASYNC_LDS 1
#  endif
#endif
#ifndef USE_ASYNC_LDS
#  define USE_ASYNC_LDS 0
#endif

__device__ __forceinline__ void copy16_to_lds(_Float16* l, const _Float16* g) {
#if USE_ASYNC_LDS
  __builtin_amdgcn_global_load_async_to_lds_b128(
      (AS1_ v4i*)g, (AS3_ v4i*)l, 0, 0);
#else
  *(v8h*)l = *(const v8h*)g;
#endif
}
__device__ __forceinline__ void lds_copy_join() {
#if USE_ASYNC_LDS
  __builtin_amdgcn_s_wait_asynccnt(0);
#endif
}

// Fragment: element h of v16h maps to k = 8*(lane>=16) + 16*(h>=8) + (h&7):
// two 16B loads at (base+koff) and (base+koff+16), koff = 8*(lane>=16).
__device__ __forceinline__ v16h make_frag(const _Float16* p) {
  v8h lo = *(const v8h*)p;
  v8h hi = *(const v8h*)(p + 16);
  v16h r;
#pragma unroll
  for (int j = 0; j < 8; ++j) { r[j] = lo[j]; r[j + 8] = hi[j]; }
  return r;
}

__device__ __forceinline__ v8f wmma_f16(v16h a, v16h b, v8f c) {
  return __builtin_amdgcn_wmma_f32_16x16x32_f16(false, a, false, b, (short)0, c, false, false);
}

// ---------------------------------------------------------------- f32 -> f16
__global__ void cvt_f32_f16(const float* __restrict__ x, _Float16* __restrict__ y, int n) {
  int i = blockIdx.x * blockDim.x + threadIdx.x;
  if (i < n) y[i] = (_Float16)x[i];
}

// ------------------------------------------- f32 [K,N] -> f16 [N,K] transpose
__global__ __launch_bounds__(256) void cvt_transpose(
    const float* __restrict__ x, _Float16* __restrict__ y, int K, int N)
{
  __shared__ float t[32][33];
  const int n0 = blockIdx.x * 32, k0 = blockIdx.y * 32;
  const int tx = threadIdx.x & 31, ty = threadIdx.x >> 5;
#pragma unroll
  for (int i = ty; i < 32; i += 8)
    t[i][tx] = x[(size_t)(k0 + i) * N + n0 + tx];
  __syncthreads();
#pragma unroll
  for (int i = ty; i < 32; i += 8)
    y[(size_t)(n0 + i) * K + k0 + tx] = (_Float16)t[tx][i];
}

// ---------------------------------------------------------------- WMMA GEMM
// C[M,N] f32 = A[M,K] f16 row-major  *  Wt[N,K] f16 row-major (B pre-transposed)
// 128 threads = 4 waves (2x2), block tile 128x128, wave tile 64x64 (4x4 frags,
// 16 WMMAs per K-slab per wave), K-slab 32, double-buffered async LDS staging.
__global__ __launch_bounds__(128) void gemm_wmma(
    const _Float16* __restrict__ A, const _Float16* __restrict__ Wt,
    float* __restrict__ C, int M, int N, int K)
{
  __shared__ __align__(16) _Float16 sA[2][128 * 32];  // 8KB per buffer
  __shared__ __align__(16) _Float16 sB[2][128 * 32];
  const int tid  = threadIdx.x;
  const int wave = tid >> 5, lane = tid & 31;
  const int wr = wave >> 1, wc = wave & 1;
  const int half = lane >> 4, l16 = lane & 15;
  const int koff = half * 8;
  const int row0 = blockIdx.y * 128, col0 = blockIdx.x * 128;

  // Each thread stages 4 x 16B of A and 4 x 16B of B per slab.
  const int r0 = tid >> 2;       // rows r0 + 32*j
  const int c0 = (tid & 3) * 8;  // k-offset (halves) within slab
  int li[4];
  const _Float16 *gA[4], *gB[4];
#pragma unroll
  for (int j = 0; j < 4; ++j) {
    li[j] = (r0 + 32 * j) * 32 + c0;
    gA[j] = A  + (size_t)(row0 + r0 + 32 * j) * K + c0;
    gB[j] = Wt + (size_t)(col0 + r0 + 32 * j) * K + c0;
  }

  v8f acc[4][4];
#pragma unroll
  for (int i = 0; i < 4; ++i)
#pragma unroll
    for (int j = 0; j < 4; ++j)
#pragma unroll
      for (int e = 0; e < 8; ++e) acc[i][j][e] = 0.0f;

  // Prologue: stage slab 0 into buffer 0.
#pragma unroll
  for (int j = 0; j < 4; ++j) {
    copy16_to_lds(&sA[0][li[j]], gA[j]);
    copy16_to_lds(&sB[0][li[j]], gB[j]);
  }
  lds_copy_join();
  __syncthreads();

  int cur = 0;
  for (int k0 = 0; k0 < K; k0 += 32) {
    const int nxt = cur ^ 1;
    if (k0 + 32 < K) {  // issue next slab; overlaps the WMMAs below
#pragma unroll
      for (int j = 0; j < 4; ++j) {
        copy16_to_lds(&sA[nxt][li[j]], gA[j] + k0 + 32);
        copy16_to_lds(&sB[nxt][li[j]], gB[j] + k0 + 32);
      }
    }

    v16h af[4], bf[4];
#pragma unroll
    for (int fm = 0; fm < 4; ++fm)
      af[fm] = make_frag(&sA[cur][(wr * 64 + fm * 16 + l16) * 32 + koff]);
#pragma unroll
    for (int fn = 0; fn < 4; ++fn)
      bf[fn] = make_frag(&sB[cur][(wc * 64 + fn * 16 + l16) * 32 + koff]);
#pragma unroll
    for (int fm = 0; fm < 4; ++fm)
#pragma unroll
      for (int fn = 0; fn < 4; ++fn)
        acc[fm][fn] = wmma_f16(af[fm], bf[fn], acc[fm][fn]);

    lds_copy_join();
    __syncthreads();
    cur = nxt;
  }

  // C layout: lanes 0-15 vgpr r -> M=r, lanes 16-31 -> M=8+r; N = l16
#pragma unroll
  for (int fm = 0; fm < 4; ++fm)
#pragma unroll
    for (int fn = 0; fn < 4; ++fn)
#pragma unroll
      for (int r = 0; r < 8; ++r) {
        int row = row0 + wr * 64 + fm * 16 + half * 8 + r;
        int col = col0 + wc * 64 + fn * 16 + l16;
        C[(size_t)row * N + col] = acc[fm][fn][r];
      }
}

// ------------------------------------------------- RMSNorm + RoPE (Q and K)
__global__ __launch_bounds__(256) void rms_rope(
    const float* __restrict__ X, const float* __restrict__ cs,
    const float* __restrict__ sn, const float* __restrict__ w,
    _Float16* __restrict__ out, int NH)
{
  __shared__ float red[256];
  __shared__ float nb[256];
  int idx  = blockIdx.x;
  int head = idx % NH;
  int l    = (idx / NH) % L_;
  int b    = idx / (NH * L_);
  int d    = threadIdx.x;

  float x = X[((size_t)(b * L_ + l) * NH + head) * DH_ + d];
  red[d] = x * x;
  __syncthreads();
  for (int st = 128; st > 0; st >>= 1) {
    if (d < st) red[d] += red[d + st];
    __syncthreads();
  }
  float n = x * rsqrtf(red[0] * (1.0f / DH_) + EPS_) * w[d];
  nb[d] = n;
  __syncthreads();
  float rot = (d < 128) ? -nb[d + 128] : nb[d - 128];
  float cv = cs[(size_t)(b * L_ + l) * DH_ + d];
  float sv = sn[(size_t)(b * L_ + l) * DH_ + d];
  out[((size_t)(b * NH + head) * L_ + l) * DH_ + d] = (_Float16)(n * cv + rot * sv);
}

// ------------------------------------------------- RMSNorm V -> [B,HKV,DH,L]
__global__ __launch_bounds__(256) void rms_v(const float* __restrict__ X,
                                             _Float16* __restrict__ vt)
{
  __shared__ float red[256];
  int idx  = blockIdx.x;
  int head = idx % HKV_;
  int l    = (idx / HKV_) % L_;
  int b    = idx / (HKV_ * L_);
  int d    = threadIdx.x;

  float x = X[((size_t)(b * L_ + l) * HKV_ + head) * DH_ + d];
  red[d] = x * x;
  __syncthreads();
  for (int st = 128; st > 0; st >>= 1) {
    if (d < st) red[d] += red[d + st];
    __syncthreads();
  }
  float n = x * rsqrtf(red[0] * (1.0f / DH_) + EPS_);
  vt[((size_t)(b * HKV_ + head) * DH_ + d) * L_ + l] = (_Float16)n;
}

// ------------------------------------------------- Flash attention (WMMA)
// Qh: [B,H,L,DH]; Kh: [B,HKV,L,DH]; Vt: [B,HKV,DH,L]; AttnH: [B*L, H*DH] (f16)
__global__ __launch_bounds__(128) void attn_wmma(
    const _Float16* __restrict__ Qh, const _Float16* __restrict__ Kh,
    const _Float16* __restrict__ Vt, _Float16* __restrict__ AttnH)
{
  __shared__ __align__(16) _Float16 sQ[64 * 256];    // 32KB [qrow][dh]
  __shared__ __align__(16) _Float16 sKV[32 * 256];   // 16KB K:[key][dh] then V:[dh][key]
  __shared__ __align__(16) _Float16 sP[4][16 * 32];  // 4KB per-wave P scratch

  const int tid  = threadIdx.x;
  const int w    = tid >> 5, lane = tid & 31;
  const int half = lane >> 4, l16 = lane & 15;
  const int koff = half * 8;
  const int t0 = blockIdx.x * 64;
  const int h  = blockIdx.y, b = blockIdx.z;
  const int hkv = h >> 1;  // GQA expand = 2

  const _Float16* Qb = Qh + ((size_t)(b * H_ + h) * L_ + t0) * DH_;
  const _Float16* Kb = Kh + (size_t)(b * HKV_ + hkv) * L_ * DH_;
  const _Float16* Vb = Vt + (size_t)(b * HKV_ + hkv) * DH_ * L_;

  // Per-thread staging geometry (all copies are 16B).
  const int qc = (tid & 31) * 8;  // dh offset (halves), Q/K tiles
  const int qr = tid >> 5;        // starting row, step 4
  const int vc = (tid & 3) * 8;   // key offset (halves), V tile
  const int vr = tid >> 2;        // starting dh row, step 32

  // Stage the whole Q tile (64 x 256 halves).
#pragma unroll
  for (int j = 0; j < 16; ++j) {
    int r = qr + 4 * j;
    copy16_to_lds(&sQ[r * 256 + qc], Qb + (size_t)r * DH_ + qc);
  }

  float m_i[8], l_i[8];
  v8f o[16];
#pragma unroll
  for (int r = 0; r < 8; ++r) { m_i[r] = -1e30f; l_i[r] = 0.0f; }
#pragma unroll
  for (int d = 0; d < 16; ++d)
#pragma unroll
    for (int e = 0; e < 8; ++e) o[d][e] = 0.0f;

  const int nkb = t0 / 32 + 2;  // causal: keys up to t0+63
  for (int kb = 0; kb < nkb; ++kb) {
    const int key0 = kb * 32;
    __syncthreads();  // sKV free (previous V reads done)
    // Stage K tile [32 keys][256 dh]
#pragma unroll
    for (int j = 0; j < 8; ++j) {
      int r = qr + 4 * j;
      copy16_to_lds(&sKV[r * 256 + qc], Kb + (size_t)(key0 + r) * DH_ + qc);
    }
    lds_copy_join();
    __syncthreads();

    // S = Q * K^T : two 16x16 accumulators (keys 0-15 / 16-31), 8 K-chunks
    v8f s0, s1;
#pragma unroll
    for (int e = 0; e < 8; ++e) { s0[e] = 0.0f; s1[e] = 0.0f; }
#pragma unroll
    for (int c = 0; c < 8; ++c) {
      v16h aq  = make_frag(&sQ[(w * 16 + l16) * 256 + c * 32 + koff]);
      v16h bk0 = make_frag(&sKV[l16 * 256 + c * 32 + koff]);
      v16h bk1 = make_frag(&sKV[(16 + l16) * 256 + c * 32 + koff]);
      s0 = wmma_f16(aq, bk0, s0);
      s1 = wmma_f16(aq, bk1, s1);
    }

    // Softcap + causal mask + online softmax (rows live in one half-wave)
#pragma unroll
    for (int r = 0; r < 8; ++r) {
      int qrow = t0 + w * 16 + half * 8 + r;
      float x0 = SOFTCAP_ * tanhf(s0[r] * (SCALING_ / SOFTCAP_));
      float x1 = SOFTCAP_ * tanhf(s1[r] * (SCALING_ / SOFTCAP_));
      if (key0 + l16 > qrow)      x0 -= 1e9f;
      if (key0 + 16 + l16 > qrow) x1 -= 1e9f;
      float mx = fmaxf(x0, x1);
#pragma unroll
      for (int offs = 8; offs >= 1; offs >>= 1)
        mx = fmaxf(mx, __shfl_xor(mx, offs, 32));
      float mnew  = fmaxf(m_i[r], mx);
      float scale = __expf(m_i[r] - mnew);
      float p0 = __expf(x0 - mnew);
      float p1 = __expf(x1 - mnew);
      float rs = p0 + p1;
#pragma unroll
      for (int offs = 8; offs >= 1; offs >>= 1)
        rs += __shfl_xor(rs, offs, 32);
      l_i[r] = l_i[r] * scale + rs;
      m_i[r] = mnew;
#pragma unroll
      for (int d = 0; d < 16; ++d) o[d][r] *= scale;
      int prow = half * 8 + r;  // re-layout P via wave-private LDS (in-order)
      sP[w][prow * 32 + l16]      = (_Float16)p0;
      sP[w][prow * 32 + 16 + l16] = (_Float16)p1;
    }
    v16h ap = make_frag(&sP[w][l16 * 32 + koff]);  // P as 16x32 A-fragment

    __syncthreads();  // all waves done with K tile in sKV
    // Stage V tile [256 dh][32 keys] (Vt is dh-major: straight copy)
#pragma unroll
    for (int j = 0; j < 8; ++j) {
      int r = vr + 32 * j;
      copy16_to_lds(&sKV[r * 32 + vc], Vb + (size_t)r * L_ + key0 + vc);
    }
    lds_copy_join();
    __syncthreads();

    // O += P * V : 16 column fragments over DH=256
#pragma unroll
    for (int d = 0; d < 16; ++d) {
      v16h bv = make_frag(&sKV[(d * 16 + l16) * 32 + koff]);
      o[d] = wmma_f16(ap, bv, o[d]);
    }
  }

  // Normalize, write f16 [B*L, H*DH]
#pragma unroll
  for (int r = 0; r < 8; ++r) {
    float inv = 1.0f / l_i[r];
    int q = t0 + w * 16 + half * 8 + r;
    size_t base = ((size_t)b * L_ + q) * (H_ * DH_) + h * DH_;
#pragma unroll
    for (int d = 0; d < 16; ++d)
      AttnH[base + d * 16 + l16] = (_Float16)(o[d][r] * inv);
  }
}

// ---------------------------------------------------------------- launch
extern "C" void kernel_launch(void* const* d_in, const int* in_sizes, int n_in,
                              void* d_out, int out_size, void* d_ws, size_t ws_size,
                              hipStream_t stream)
{
  (void)in_sizes; (void)n_in; (void)out_size; (void)ws_size;
  const float* hs   = (const float*)d_in[0];
  const float* cosb = (const float*)d_in[1];
  const float* sinb = (const float*)d_in[2];
  // d_in[3] attention_mask: causal mask applied analytically (same values)
  const float* Wq = (const float*)d_in[4];
  const float* Wk = (const float*)d_in[5];
  const float* Wv = (const float*)d_in[6];
  const float* Wo = (const float*)d_in[7];
  const float* qw = (const float*)d_in[8];
  const float* kw = (const float*)d_in[9];
  float* out = (float*)d_out;

  char* ws = (char*)d_ws;
  size_t off = 0;
  auto alloc = [&](size_t bytes) -> char* {
    char* p = ws + off;
    off += (bytes + 255) & ~(size_t)255;
    return p;
  };

  const int M = B_ * L_;  // 4096
  _Float16* Xh  = (_Float16*)alloc((size_t)M * HS_ * 2);
  _Float16* Wqt = (_Float16*)alloc((size_t)HS_ * H_ * DH_ * 2);    // [4096, 2560]
  _Float16* Wkt = (_Float16*)alloc((size_t)HS_ * HKV_ * DH_ * 2);  // [2048, 2560]
  _Float16* Wvt = (_Float16*)alloc((size_t)HS_ * HKV_ * DH_ * 2);
  _Float16* Wot = (_Float16*)alloc((size_t)H_ * DH_ * HS_ * 2);    // [2560, 4096]
  float*    Qf  = (float*)alloc((size_t)M * H_ * DH_ * 4);
  float*    Kf  = (float*)alloc((size_t)M * HKV_ * DH_ * 4);
  float*    Vf  = (float*)alloc((size_t)M * HKV_ * DH_ * 4);
  _Float16* Qh  = (_Float16*)alloc((size_t)B_ * H_ * L_ * DH_ * 2);
  _Float16* Kh  = (_Float16*)alloc((size_t)B_ * HKV_ * L_ * DH_ * 2);
  _Float16* Vt  = (_Float16*)alloc((size_t)B_ * HKV_ * DH_ * L_ * 2);
  _Float16* Ah  = (_Float16*)alloc((size_t)M * H_ * DH_ * 2);

  // Convert activations; convert+transpose weights (once) so all GEMM staging
  // is contiguous b128 (async-to-LDS eligible).
  cvt_f32_f16<<<(M * HS_ + 255) / 256, 256, 0, stream>>>(hs, Xh, M * HS_);
  cvt_transpose<<<dim3(H_ * DH_ / 32, HS_ / 32), 256, 0, stream>>>(Wq, Wqt, HS_, H_ * DH_);
  cvt_transpose<<<dim3(HKV_ * DH_ / 32, HS_ / 32), 256, 0, stream>>>(Wk, Wkt, HS_, HKV_ * DH_);
  cvt_transpose<<<dim3(HKV_ * DH_ / 32, HS_ / 32), 256, 0, stream>>>(Wv, Wvt, HS_, HKV_ * DH_);
  cvt_transpose<<<dim3(HS_ / 32, H_ * DH_ / 32), 256, 0, stream>>>(Wo, Wot, H_ * DH_, HS_);

  // Projections (block tile 128x128)
  gemm_wmma<<<dim3(H_ * DH_ / 128, M / 128), 128, 0, stream>>>(Xh, Wqt, Qf, M, H_ * DH_, HS_);
  gemm_wmma<<<dim3(HKV_ * DH_ / 128, M / 128), 128, 0, stream>>>(Xh, Wkt, Kf, M, HKV_ * DH_, HS_);
  gemm_wmma<<<dim3(HKV_ * DH_ / 128, M / 128), 128, 0, stream>>>(Xh, Wvt, Vf, M, HKV_ * DH_, HS_);

  // Norm + RoPE, layouts for attention
  rms_rope<<<B_ * L_ * H_,   256, 0, stream>>>(Qf, cosb, sinb, qw, Qh, H_);
  rms_rope<<<B_ * L_ * HKV_, 256, 0, stream>>>(Kf, cosb, sinb, kw, Kh, HKV_);
  rms_v  <<<B_ * L_ * HKV_,  256, 0, stream>>>(Vf, Vt);

  // Attention
  attn_wmma<<<dim3(L_ / 64, H_, B_), 128, 0, stream>>>(Qh, Kh, Vt, Ah);

  // Output projection -> f32 result
  gemm_wmma<<<dim3(HS_ / 128, M / 128), 128, 0, stream>>>(Ah, Wot, out, M, HS_, H_ * DH_);
}
